// Transformer_88837103550607
// MI455X (gfx1250) — matrix operand order repaired and checked by reading, not compiled, not run on hardware
//
#include <hip/hip_runtime.h>

typedef __attribute__((ext_vector_type(16))) _Float16 v16h;
typedef __attribute__((ext_vector_type(8)))  _Float16 v8h;
typedef __attribute__((ext_vector_type(8)))  float    v8f;
typedef __attribute__((ext_vector_type(4)))  float    vf4;

#define B_     32
#define T_     96
#define D_     512
#define DM_    4
#define DIM_   512
#define DEPTH_ 4
#define H_     8
#define DH_    64
#define MLP_   2048
#define INNER_ 512
#define QKVW   1536
#define NTOK   516            // D + DM variate tokens
#define NPAD   528            // padded row count (multiple of 16)
#define MTOT   (B_ * NPAD)    // 16896 total GEMM rows
#define NKPAD  544            // key padding (multiple of 32) for PV K-loop

#define BM 128
#define BN 128
#define BK 64                 // two WMMA k-steps per staged tile
#define LDSPK 72              // padded LDS row stride in halfs (144B, 16B-aligned)
#define EBK 32                // embed kernel k-tile
#define LDSP 40               // embed LDS row stride in halfs

union FragH { v16h v; vf4 f4[2]; };

__device__ __forceinline__ void cvt4h(_Float16* d, vf4 f) {
  d[0] = (_Float16)f[0]; d[1] = (_Float16)f[1];
  d[2] = (_Float16)f[2]; d[3] = (_Float16)f[3];
}

// CDNA5 async global->LDS copy (ASYNCcnt-tracked), 16B per lane.
__device__ __forceinline__ void async_ld_b128(unsigned lds_off, const void* g) {
  asm volatile("global_load_async_to_lds_b128 %0, %1, off"
               :: "v"(lds_off), "v"((unsigned long long)(size_t)g)
               : "memory");
}
__device__ __forceinline__ void wait_async_le0() {
  asm volatile("s_wait_asynccnt 0" ::: "memory");
}
__device__ __forceinline__ void wait_async_le8() {
  asm volatile("s_wait_asynccnt 8" ::: "memory");
}

// ---------------------------------------------------------------------------
// f32 -> f16 tensor conversion (weights, once per launch)
// ---------------------------------------------------------------------------
__global__ void cvt_f32_f16_kernel(const float* __restrict__ s,
                                   _Float16* __restrict__ d, long n) {
  const long i = (long)blockIdx.x * 256 + threadIdx.x;
  if (i < n) d[i] = (_Float16)s[i];
}

// ---------------------------------------------------------------------------
// f16 WMMA GEMM: out[M,N] = A[M,K](f16) @ Bw[N,K](f16)^T (+bias)(+gelu|+resid)
// Double-buffered async global->LDS pipeline: stage i+1's DMA runs behind
// stage i's 16 WMMAs; s_wait_asynccnt 8 retires only the older stage.
// EPI 0: (+bias)  1: (+bias, exact GELU)  2: (+bias, + resid f32)
// OUT16 1: store f16 to outh  0: store f32 to outf
// ---------------------------------------------------------------------------
template<int EPI, int OUT16>
__global__ void __launch_bounds__(256) gemm_f16(
    const _Float16* __restrict__ A, const _Float16* __restrict__ Bw,
    const float* __restrict__ bias, const float* __restrict__ resid,
    float* __restrict__ outf, _Float16* __restrict__ outh, int N, int K)
{
  __shared__ __align__(16) _Float16 lds_a[2][BM][LDSPK];
  __shared__ __align__(16) _Float16 lds_b[2][BN][LDSPK];

  const int tid = threadIdx.x, lane = tid & 31, wave = tid >> 5;
  const int waveM = wave >> 1, waveN = wave & 1;       // 4 x 2 wave grid
  const int m0 = blockIdx.y * BM, n0 = blockIdx.x * BN;
  const int row2 = tid >> 1, seg = (tid & 1) * 32;     // 128 rows x 2 half-rows
  const int mr = lane & 15, hi = lane >> 4, kb = hi * 8, kh = hi * 16;

  const _Float16* ag = A  + (long)(m0 + row2) * K + seg;
  const _Float16* bg = Bw + (long)(n0 + row2) * K + seg;
  const unsigned la[2] = { (unsigned)(size_t)&lds_a[0][row2][seg],
                           (unsigned)(size_t)&lds_a[1][row2][seg] };
  const unsigned lb[2] = { (unsigned)(size_t)&lds_b[0][row2][seg],
                           (unsigned)(size_t)&lds_b[1][row2][seg] };

  v8f acc[2][4] = {};
  const int nk = K / BK;

  // prologue: stage 0
  #pragma unroll
  for (int t = 0; t < 4; ++t) {
    async_ld_b128(la[0] + t * 16, ag + t * 8);
    async_ld_b128(lb[0] + t * 16, bg + t * 8);
  }

  for (int i = 0; i < nk; ++i) {
    const int cur = i & 1;
    if (i + 1 < nk) {
      const _Float16* an = ag + (i + 1) * BK;
      const _Float16* bn = bg + (i + 1) * BK;
      #pragma unroll
      for (int t = 0; t < 4; ++t) {
        async_ld_b128(la[cur ^ 1] + t * 16, an + t * 8);
        async_ld_b128(lb[cur ^ 1] + t * 16, bn + t * 8);
      }
      wait_async_le8();      // stage i retired (in-order), stage i+1 in flight
    } else {
      wait_async_le0();
    }
    __syncthreads();

    #pragma unroll
    for (int s = 0; s < 2; ++s) {          // two 16x16x32 k-steps
      FragH af[2], bf[4];
      #pragma unroll
      for (int sm = 0; sm < 2; ++sm) {
        const _Float16* p = &lds_a[cur][waveM * 32 + sm * 16 + mr][s * 32];
        af[sm].f4[0] = *(const vf4*)(p + kb);
        af[sm].f4[1] = *(const vf4*)(p + 16 + kb);
      }
      #pragma unroll
      for (int sn = 0; sn < 4; ++sn) {
        const _Float16* p = &lds_b[cur][waveN * 64 + sn * 16 + mr][s * 32];
        bf[sn].f4[0] = *(const vf4*)(p + kb);
        bf[sn].f4[1] = *(const vf4*)(p + 16 + kb);
      }
      #pragma unroll
      for (int sm = 0; sm < 2; ++sm)
        #pragma unroll
        for (int sn = 0; sn < 4; ++sn)
          acc[sm][sn] = __builtin_amdgcn_wmma_f32_16x16x32_f16(
              false, af[sm].v, false, bf[sn].v, (short)0, acc[sm][sn], false, false);
    }
    __syncthreads();
  }

  #pragma unroll
  for (int sm = 0; sm < 2; ++sm) {
    #pragma unroll
    for (int sn = 0; sn < 4; ++sn) {
      const int col = n0 + waveN * 64 + sn * 16 + mr;
      const float bv = bias ? bias[col] : 0.0f;
      #pragma unroll
      for (int i = 0; i < 8; ++i) {
        const int row = m0 + waveM * 32 + sm * 16 + hi * 8 + i;
        float v = acc[sm][sn][i] + bv;
        if (EPI == 1) v = 0.5f * v * (1.0f + erff(v * 0.70710678118654752f));
        if (EPI == 2) v += resid[(long)row * N + col];
        if (OUT16) outh[(long)row * N + col] = (_Float16)v;
        else       outf[(long)row * N + col] = v;
      }
    }
  }
}

// ---------------------------------------------------------------------------
// Embedding GEMM: h[M,512] = concat(x,x_mark)^T @ W_emb^T + b_emb  (K = 96)
// A gathered+converted from f32 inputs; B (f16 weights) via async-to-LDS.
// ---------------------------------------------------------------------------
__global__ void __launch_bounds__(256) embed_gemm(
    const float* __restrict__ xg, const float* __restrict__ xmg,
    const _Float16* __restrict__ Wh, const float* __restrict__ bias,
    float* __restrict__ out)
{
  __shared__ __align__(16) _Float16 lds_a[BM][LDSP];
  __shared__ __align__(16) _Float16 lds_b[BN][LDSP];

  const int tid = threadIdx.x, lane = tid & 31, wave = tid >> 5;
  const int waveM = wave >> 1, waveN = wave & 1;
  const int m0 = blockIdx.y * BM, n0 = blockIdx.x * BN;
  const int row2 = tid >> 1, seg = (tid & 1) * 16;
  const int mr = lane & 15, hi = lane >> 4, kb = hi * 8, kh = hi * 16;

  const int grow = m0 + row2;
  const int eb = grow / NPAD, en = grow % NPAD;
  const _Float16* bg = Wh + (long)(n0 + row2) * T_ + seg;
  const unsigned lb = (unsigned)(size_t)&lds_b[row2][seg];

  v8f acc[2][4] = {};

  for (int k0 = 0; k0 < T_; k0 += EBK) {
    async_ld_b128(lb,      bg + k0);
    async_ld_b128(lb + 16, bg + k0 + 8);
    #pragma unroll 4
    for (int c = 0; c < 16; ++c) {
      const int k = k0 + seg + c;              // time index t < 96
      float v = 0.0f;
      if (en < D_)        v = xg [((long)eb * T_ + k) * D_  + en];
      else if (en < NTOK) v = xmg[((long)eb * T_ + k) * DM_ + (en - D_)];
      lds_a[row2][seg + c] = (_Float16)v;
    }
    wait_async_le0();
    __syncthreads();

    FragH af[2], bf[4];
    #pragma unroll
    for (int sm = 0; sm < 2; ++sm) {
      const _Float16* p = &lds_a[waveM * 32 + sm * 16 + mr][0];
      af[sm].f4[0] = *(const vf4*)(p + kb);
      af[sm].f4[1] = *(const vf4*)(p + 16 + kb);
    }
    #pragma unroll
    for (int sn = 0; sn < 4; ++sn) {
      const _Float16* p = &lds_b[waveN * 64 + sn * 16 + mr][0];
      bf[sn].f4[0] = *(const vf4*)(p + kb);
      bf[sn].f4[1] = *(const vf4*)(p + 16 + kb);
    }
    #pragma unroll
    for (int sm = 0; sm < 2; ++sm)
      #pragma unroll
      for (int sn = 0; sn < 4; ++sn)
        acc[sm][sn] = __builtin_amdgcn_wmma_f32_16x16x32_f16(
            false, af[sm].v, false, bf[sn].v, (short)0, acc[sm][sn], false, false);
    __syncthreads();
  }

  #pragma unroll
  for (int sm = 0; sm < 2; ++sm) {
    #pragma unroll
    for (int sn = 0; sn < 4; ++sn) {
      const int col = n0 + waveN * 64 + sn * 16 + mr;
      const float bv = bias[col];
      #pragma unroll
      for (int i = 0; i < 8; ++i) {
        const int row = m0 + waveM * 32 + sm * 16 + hi * 8 + i;
        out[(long)row * DIM_ + col] = acc[sm][sn][i] + bv;
      }
    }
  }
}

// ---------------------------------------------------------------------------
// Fused attention: one workgroup (4 waves) per (b, h). qkv is f16.
// K rows async-copied to LDS; V^T transposed in; per-wave 16x544 f32 scores.
// ---------------------------------------------------------------------------
__global__ void __launch_bounds__(128) attn_kernel(
    const _Float16* __restrict__ qkv, _Float16* __restrict__ o)
{
  extern __shared__ char smem[];
  _Float16* Kl = (_Float16*)smem;                    // [NKPAD][DH]
  _Float16* Vt = Kl + NKPAD * DH_;                   // [DH][NKPAD]
  float*    Sc = (float*)(Vt + DH_ * NKPAD);         // [4][16][NKPAD]
  float* rowsum = Sc + 4 * 16 * NKPAD;               // [4][16]

  const int tid  = threadIdx.x;
  const int lane = tid & 31;
  const int wave = tid >> 5;
  const int hh   = blockIdx.x;
  const int bb   = blockIdx.y;
  const int mr   = lane & 15;
  const int hi   = lane >> 4;
  const int kb   = hi * 8;
  const int kh   = hi * 16;

  // ---- K via async-to-LDS; V^T transposed manually; zero-pad [528,544) ----
  for (int r = tid; r < NKPAD; r += 128) {
    if (r < NPAD) {
      const _Float16* gk = qkv + ((long)(bb * NPAD + r)) * QKVW + INNER_ + hh * DH_;
      const unsigned ko = (unsigned)(size_t)&Kl[r * DH_];
      #pragma unroll
      for (int d0 = 0; d0 < DH_ * 2; d0 += 16)       // bytes
        async_ld_b128(ko + d0, (const char*)gk + d0);
      const _Float16* gv = gk + INNER_;
      #pragma unroll
      for (int d0 = 0; d0 < DH_; d0 += 8) {
        v8h vv = *(const v8h*)(gv + d0);
        #pragma unroll
        for (int j = 0; j < 8; ++j) Vt[(d0 + j) * NKPAD + r] = vv[j];
      }
    } else {
      v8h zz = {};
      #pragma unroll
      for (int d0 = 0; d0 < DH_; d0 += 8) *(v8h*)&Kl[r * DH_ + d0] = zz;
      for (int d = 0; d < DH_; ++d) Vt[d * NKPAD + r] = (_Float16)0.0f;
    }
  }
  wait_async_le0();
  __syncthreads();

  float* myS = Sc + wave * 16 * NKPAD;

  for (int qt = wave; qt < NPAD / 16; qt += 4) {
    // ---- Q fragments: direct f16 global b128 loads ----
    FragH qf[2];
    {
      const _Float16* qp = qkv + ((long)(bb * NPAD + qt * 16 + mr)) * QKVW + hh * DH_;
      #pragma unroll
      for (int s = 0; s < 2; ++s) {
        qf[s].f4[0] = *(const vf4*)(qp + s * 32 + kb);
        qf[s].f4[1] = *(const vf4*)(qp + s * 32 + 16 + kb);
      }
    }
    // ---- scores: S = (Q @ K^T) * scale, masked past NTOK ----
    for (int nt = 0; nt < NKPAD / 16; ++nt) {
      v8f acc = {};
      #pragma unroll
      for (int s = 0; s < 2; ++s) {
        FragH bf;
        const _Float16* p = Kl + (nt * 16 + mr) * DH_ + s * 32 + kb;
        bf.f4[0] = *(const vf4*)p;
        bf.f4[1] = *(const vf4*)(p + 16);
        acc = __builtin_amdgcn_wmma_f32_16x16x32_f16(
            false, qf[s].v, false, bf.v, (short)0, acc, false, false);
      }
      const int col = nt * 16 + mr;
      #pragma unroll
      for (int i = 0; i < 8; ++i) {
        float sv = acc[i] * 0.125f;            // DH^-0.5
        if (col >= NTOK) sv = -1e30f;
        myS[(hi * 8 + i) * NKPAD + col] = sv;
      }
    }
    // ---- softmax (unnormalized exp + row sums); 2 lanes per row ----
    {
      const int row = mr;
      const int c0  = hi * (NKPAD / 2);
      float mx = -1e30f;
      for (int c = 0; c < NKPAD / 2; ++c)
        mx = fmaxf(mx, myS[row * NKPAD + c0 + c]);
      mx = fmaxf(mx, __shfl_xor(mx, 16, 32));
      float sum = 0.0f;
      for (int c = 0; c < NKPAD / 2; ++c) {
        float e = __expf(myS[row * NKPAD + c0 + c] - mx);
        myS[row * NKPAD + c0 + c] = e;
        sum += e;
      }
      sum += __shfl_xor(sum, 16, 32);
      if (hi == 0) rowsum[wave * 16 + row] = sum;
    }
    // ---- O = S_exp @ V, normalized at writeout (f16) ----
    #pragma unroll
    for (int dt = 0; dt < 4; ++dt) {
      v8f acc = {};
      for (int kt = 0; kt < NKPAD / 32; ++kt) {
        FragH af;
        const float* sp = myS + mr * NKPAD + kt * 32;
        vf4 a0 = *(const vf4*)(sp + kb);
        vf4 a1 = *(const vf4*)(sp + kb + 4);
        vf4 a2 = *(const vf4*)(sp + 16 + kb);
        vf4 a3 = *(const vf4*)(sp + 16 + kb + 4);
        _Float16* ah = (_Float16*)&af;
        cvt4h(ah, a0); cvt4h(ah + 4, a1); cvt4h(ah + 8, a2); cvt4h(ah + 12, a3);
        FragH bf;
        const _Float16* vp = Vt + (dt * 16 + mr) * NKPAD + kt * 32 + kb;
        bf.f4[0] = *(const vf4*)vp;
        bf.f4[1] = *(const vf4*)(vp + 16);
        acc = __builtin_amdgcn_wmma_f32_16x16x32_f16(
            false, af.v, false, bf.v, (short)0, acc, false, false);
      }
      #pragma unroll
      for (int i = 0; i < 8; ++i) {
        const int rloc = hi * 8 + i;
        const float inv = 1.0f / rowsum[wave * 16 + rloc];
        const int row = qt * 16 + rloc;
        o[((long)(bb * NPAD + row)) * INNER_ + hh * DH_ + dt * 16 + mr] =
            (_Float16)(acc[i] * inv);
      }
    }
  }
}

// ---------------------------------------------------------------------------
// LayerNorm: one wave per 512-wide f32 row, f16 output
// ---------------------------------------------------------------------------
__global__ void __launch_bounds__(256) ln_kernel(
    const float* __restrict__ x, const float* __restrict__ g,
    const float* __restrict__ b, _Float16* __restrict__ z)
{
  const int lane = threadIdx.x & 31;
  const int wave = threadIdx.x >> 5;
  const long row = (long)blockIdx.x * 8 + wave;
  const float* xp = x + row * DIM_;
  float v[16], s = 0.0f, sq = 0.0f;
  #pragma unroll
  for (int j = 0; j < 16; ++j) {
    v[j] = xp[lane + 32 * j];
    s += v[j]; sq += v[j] * v[j];
  }
  #pragma unroll
  for (int o = 16; o >= 1; o >>= 1) {
    s  += __shfl_xor(s,  o, 32);
    sq += __shfl_xor(sq, o, 32);
  }
  const float mean = s * (1.0f / DIM_);
  const float var  = sq * (1.0f / DIM_) - mean * mean;
  const float rstd = rsqrtf(var + 1e-5f);
  _Float16* zp = z + row * DIM_;
  #pragma unroll
  for (int j = 0; j < 16; ++j) {
    const int c = lane + 32 * j;
    zp[c] = (_Float16)((v[j] - mean) * rstd * g[c] + b[c]);
  }
}

// ---------------------------------------------------------------------------
// Strip row padding into the output buffer
// ---------------------------------------------------------------------------
__global__ void copy_out_kernel(const float* __restrict__ h, float* __restrict__ out)
{
  const long i = (long)blockIdx.x * 256 + threadIdx.x;
  const long tot = (long)B_ * NTOK * DIM_;
  if (i >= tot) return;
  const long c = i % DIM_;
  const long r = i / DIM_;
  const long n = r % NTOK;
  const long b = r / NTOK;
  out[i] = h[((b * NPAD + n) * DIM_) + c];
}

extern "C" void kernel_launch(void* const* d_in, const int* in_sizes, int n_in,
                              void* d_out, int out_size, void* d_ws, size_t ws_size,
                              hipStream_t stream) {
  (void)in_sizes; (void)n_in; (void)out_size; (void)ws_size;
  const float* x     = (const float*)d_in[0];
  const float* xm    = (const float*)d_in[1];
  const float* W_emb = (const float*)d_in[2];
  const float* b_emb = (const float*)d_in[3];
  const float* ln1g  = (const float*)d_in[4];
  const float* ln1b  = (const float*)d_in[5];
  const float* Wqkv  = (const float*)d_in[6];
  const float* Wo    = (const float*)d_in[7];
  const float* bo    = (const float*)d_in[8];
  const float* ln2g  = (const float*)d_in[9];
  const float* ln2b  = (const float*)d_in[10];
  const float* W1    = (const float*)d_in[11];
  const float* b1    = (const float*)d_in[12];
  const float* W2    = (const float*)d_in[13];
  const float* b2    = (const float*)d_in[14];
  float* out = (float*)d_out;

  char* ws = (char*)d_ws;
  size_t off = 0;
  float*    h    = (float*)(ws + off);    off += (size_t)MTOT * DIM_ * 4;
  _Float16* z    = (_Float16*)(ws + off); off += (size_t)MTOT * DIM_ * 2;
  _Float16* o    = (_Float16*)(ws + off); off += (size_t)MTOT * DIM_ * 2;
  _Float16* big  = (_Float16*)(ws + off); off += (size_t)MTOT * MLP_ * 2;   // qkv/ff
  _Float16* Weh  = (_Float16*)(ws + off); off += (size_t)DIM_ * T_ * 2;
  _Float16* Wqh  = (_Float16*)(ws + off); off += (size_t)DEPTH_ * QKVW * DIM_ * 2;
  _Float16* Woh  = (_Float16*)(ws + off); off += (size_t)DEPTH_ * DIM_ * INNER_ * 2;
  _Float16* W1h  = (_Float16*)(ws + off); off += (size_t)DEPTH_ * MLP_ * DIM_ * 2;
  _Float16* W2h  = (_Float16*)(ws + off); off += (size_t)DEPTH_ * DIM_ * MLP_ * 2;

  // weight conversions (f32 -> f16), once per launch
  {
    long n;
    n = (long)DIM_ * T_;
    cvt_f32_f16_kernel<<<(n + 255) / 256, 256, 0, stream>>>(W_emb, Weh, n);
    n = (long)DEPTH_ * QKVW * DIM_;
    cvt_f32_f16_kernel<<<(n + 255) / 256, 256, 0, stream>>>(Wqkv, Wqh, n);
    n = (long)DEPTH_ * DIM_ * INNER_;
    cvt_f32_f16_kernel<<<(n + 255) / 256, 256, 0, stream>>>(Wo, Woh, n);
    n = (long)DEPTH_ * MLP_ * DIM_;
    cvt_f32_f16_kernel<<<(n + 255) / 256, 256, 0, stream>>>(W1, W1h, n);
    n = (long)DEPTH_ * DIM_ * MLP_;
    cvt_f32_f16_kernel<<<(n + 255) / 256, 256, 0, stream>>>(W2, W2h, n);
  }

  const size_t smem = (size_t)NKPAD * DH_ * 2     // K (f16)
                    + (size_t)DH_ * NKPAD * 2     // V^T (f16)
                    + (size_t)4 * 16 * NKPAD * 4  // score strips (f32)
                    + (size_t)4 * 16 * 4;         // row sums

  // Embedding: h = concat(x,x_mark)^T @ W_emb^T + b_emb
  embed_gemm<<<dim3(DIM_ / BN, MTOT / BM), 256, 0, stream>>>(x, xm, Weh, b_emb, h);

  for (int d = 0; d < DEPTH_; ++d) {
    ln_kernel<<<MTOT / 8, 256, 0, stream>>>(h, ln1g + d * DIM_, ln1b + d * DIM_, z);
    gemm_f16<0, 1><<<dim3(QKVW / BN, MTOT / BM), 256, 0, stream>>>(
        z, Wqh + (size_t)d * QKVW * DIM_, nullptr, nullptr,
        nullptr, big, QKVW, DIM_);
    attn_kernel<<<dim3(H_, B_), 128, smem, stream>>>(big, o);
    gemm_f16<2, 0><<<dim3(DIM_ / BN, MTOT / BM), 256, 0, stream>>>(
        o, Woh + (size_t)d * DIM_ * INNER_, bo + d * DIM_, h,
        h, nullptr, DIM_, INNER_);
    ln_kernel<<<MTOT / 8, 256, 0, stream>>>(h, ln2g + d * DIM_, ln2b + d * DIM_, z);
    gemm_f16<1, 1><<<dim3(MLP_ / BN, MTOT / BM), 256, 0, stream>>>(
        z, W1h + (size_t)d * MLP_ * DIM_, b1 + d * MLP_, nullptr,
        nullptr, big, MLP_, DIM_);
    gemm_f16<2, 0><<<dim3(DIM_ / BN, MTOT / BM), 256, 0, stream>>>(
        big, W2h + (size_t)d * DIM_ * MLP_, b2 + d * DIM_, h,
        h, nullptr, DIM_, MLP_);
  }

  const long tot = (long)B_ * NTOK * DIM_;
  copy_out_kernel<<<(tot + 255) / 256, 256, 0, stream>>>(h, out);
}
